// FLARetNet_89498528514091
// MI455X (gfx1250) — matrix operand order repaired and checked by reading, not compiled, not run on hardware
//
#include <hip/hip_runtime.h>
#include <hip/hip_bf16.h>
#include <math.h>

typedef _Float16 half_t;
typedef __attribute__((ext_vector_type(16))) _Float16 v16h;
typedef __attribute__((ext_vector_type(8)))  _Float16 v8h;
typedef __attribute__((ext_vector_type(8)))  float    v8f;

static constexpr int Bb = 2, Tt = 4096, Dd = 1024, Hh = 16;
static constexpr int DK = 64, DV = 128, Cc = 64, NC = Tt / Cc;
static constexpr int MTOK = Bb * Tt; // 8192

union HV16 { v16h v; v8h h[2]; };

__device__ __forceinline__ v8f wmma16(v16h a, v16h b, v8f c) {
  return __builtin_amdgcn_wmma_f32_16x16x32_f16(false, a, false, b, (short)0, c,
                                                false, false);
}

// ---- Tensor Data Mover: 2-D tile global->LDS with row padding ----------
#if __has_builtin(__builtin_amdgcn_tensor_load_to_lds) && \
    __has_builtin(__builtin_amdgcn_s_wait_tensorcnt)
#define USE_TDM 1
typedef unsigned int u32x4 __attribute__((ext_vector_type(4)));
typedef int i32x4 __attribute__((ext_vector_type(4)));
typedef int i32x8 __attribute__((ext_vector_type(8)));

// Copy `rows` x `dim0` f16 tile (global row stride = `stride` halves) into
// LDS at lds_off, inserting pad per D# pad codes (interval/amount, ISA 8.4).
__device__ __forceinline__ void tdm_load_2d(const half_t* g, unsigned lds_off,
                                            int dim0, int rows, int stride,
                                            int padi, int pada) {
  unsigned long long ga = (unsigned long long)(size_t)g;
  u32x4 g0 = {1u,  // count=1, is_restore=0, gather off
              lds_off, (unsigned)ga,
              (unsigned)((ga >> 32) & 0x1FFFFFFull) | 0x80000000u};  // type=2
  i32x8 g1 = {
      (1 << 16) | (1 << 20) | (padi << 22) | (pada << 25),  // 2B, pad_enable
      (dim0 & 0xFFFF) << 16,                                // tensor_dim0 lo
      ((dim0 >> 16) & 0xFFFF) | ((rows & 0xFFFF) << 16),    // td0 hi | td1 lo
      ((rows >> 16) & 0xFFFF) | ((dim0 & 0xFFFF) << 16),    // td1 hi | tile0
      rows & 0xFFFF,                                        // tile1 | tile2=0
      stride,                                               // dim0 stride lo
      0, 0};
  i32x4 gz = {0, 0, 0, 0};
#if defined(__clang_major__) && __clang_major__ >= 23
  i32x8 gz8 = {0, 0, 0, 0, 0, 0, 0, 0};
  __builtin_amdgcn_tensor_load_to_lds(g0, g1, gz, gz, gz8, 0);
#else
  __builtin_amdgcn_tensor_load_to_lds(g0, g1, gz, gz, 0);
#endif
}
#define TDM_WAIT() __builtin_amdgcn_s_wait_tensorcnt(0)
#else
#define TDM_WAIT()
#endif

// ---- fragment loaders: each is 2x contiguous 16B LDS loads -------------
__device__ __forceinline__ v16h frag_a(const half_t* base, int ld, int row0,
                                       int k0, int lane) {
  const half_t* p =
      base + (size_t)(row0 + (lane & 15)) * ld + k0 + ((lane >> 4) * 8);
  HV16 u;
  u.h[0] = *(const v8h*)p;
  u.h[1] = *(const v8h*)(p + 16);
  return u.v;
}
__device__ __forceinline__ v16h frag_bT(const half_t* base, int ld, int k0,
                                        int col0, int lane) {
  const half_t* p =
      base + (size_t)(col0 + (lane & 15)) * ld + k0 + ((lane >> 4) * 8);
  HV16 u;
  u.h[0] = *(const v8h*)p;
  u.h[1] = *(const v8h*)(p + 16);
  return u.v;
}
__device__ __forceinline__ v8f frag_cT_scaled(const half_t* base, int ld,
                                              int d0, int e0, float sc,
                                              int lane) {
  const half_t* p =
      base + (size_t)(e0 + (lane & 15)) * ld + d0 + ((lane >> 4) * 8);
  v8h hv = *(const v8h*)p;
  v8f c;
#pragma unroll
  for (int j = 0; j < 8; ++j) c[j] = (float)hv[j] * sc;
  return c;
}
__device__ __forceinline__ void frag_cT_store(half_t* base, int ld, int d0,
                                              int e0, v8f c, int lane) {
  v8h hv;
#pragma unroll
  for (int j = 0; j < 8; ++j) hv[j] = (half_t)c[j];
  *(v8h*)(base + (size_t)(e0 + (lane & 15)) * ld + d0 + ((lane >> 4) * 8)) = hv;
}

// ---- f32 -> f16 conversion --------------------------------------------
__global__ void cvt_kernel(const float* __restrict__ in,
                           half_t* __restrict__ out, int n) {
  int i = blockIdx.x * blockDim.x + threadIdx.x;
  if (i < n) out[i] = (half_t)in[i];
}

// ---- tiled WMMA GEMM: C[M,N] = A[M,K] * B[K,N] -------------------------
template <bool OUT_HALF>
__global__ __launch_bounds__(256) void gemm_f16_kernel(
    const half_t* __restrict__ A, const half_t* __restrict__ Bm, void* Cout,
    int M, int N, int K) {
  __shared__ half_t sA[128 * 40];   // A tile row-major [128][32+pad8]
  __shared__ half_t sBt[128 * 40];  // B tile TRANSPOSED [n:128][k:32+pad8]
  const int tid = threadIdx.x, lane = tid & 31, wid = tid >> 5;
  const int m0b = blockIdx.y * 128, n0b = blockIdx.x * 128;
  const int wm = (wid >> 1) * 32, wn = (wid & 1) * 64;

  v8f acc[2][4];
#pragma unroll
  for (int i = 0; i < 2; ++i)
#pragma unroll
    for (int j = 0; j < 4; ++j) acc[i][j] = {};

  for (int k0 = 0; k0 < K; k0 += 32) {
#ifdef USE_TDM
    if (tid < 32)  // wave 0 issues the A-tile descriptor (rows of 32 halves =
                   // 16 DWORDs -> pad code 3; pad 8 halves = 4 DWORDs -> 3)
      tdm_load_2d(A + (size_t)m0b * K + k0, (unsigned)(size_t)sA, 32, 128, K,
                  3, 3);
#else
    {  // stage A tile (row-major copy, 16B vector ops)
      int r = tid >> 1, c0 = (tid & 1) * 16;
      const half_t* src = A + (size_t)(m0b + r) * K + k0 + c0;
      *(v8h*)&sA[r * 40 + c0] = *(const v8h*)src;
      *(v8h*)&sA[r * 40 + c0 + 8] = *(const v8h*)(src + 8);
      if (k0 + 32 < K) __builtin_prefetch(src + 32, 0, 1);
    }
#endif
    {  // stage B tile transposed
      int r = tid >> 3, c0 = (tid & 7) * 16;
      const half_t* src = Bm + (size_t)(k0 + r) * N + n0b + c0;
      v8h b0 = *(const v8h*)src;
      v8h b1 = *(const v8h*)(src + 8);
      if (k0 + 32 < K) __builtin_prefetch(src + 32 * (size_t)N, 0, 1);
#pragma unroll
      for (int u = 0; u < 8; ++u) {
        sBt[(c0 + u) * 40 + r] = b0[u];
        sBt[(c0 + 8 + u) * 40 + r] = b1[u];
      }
    }
    TDM_WAIT();
    __syncthreads();
    v16h a0 = frag_a(sA, 40, wm, 0, lane);
    v16h a1 = frag_a(sA, 40, wm + 16, 0, lane);
#pragma unroll
    for (int t = 0; t < 4; ++t) {
      v16h b = frag_bT(sBt, 40, 0, wn + 16 * t, lane);
      acc[0][t] = wmma16(a0, b, acc[0][t]);
      acc[1][t] = wmma16(a1, b, acc[1][t]);
    }
    __syncthreads();
  }
  const int rb = (lane >> 4) * 8, cl = lane & 15;
  if (OUT_HALF) {
    half_t* C = (half_t*)Cout;
#pragma unroll
    for (int tr = 0; tr < 2; ++tr)
#pragma unroll
      for (int t = 0; t < 4; ++t)
#pragma unroll
        for (int j = 0; j < 8; ++j)
          C[(size_t)(m0b + wm + tr * 16 + rb + j) * N +
            (n0b + wn + 16 * t + cl)] = (half_t)acc[tr][t][j];
  } else {
    float* C = (float*)Cout;
#pragma unroll
    for (int tr = 0; tr < 2; ++tr)
#pragma unroll
      for (int t = 0; t < 4; ++t)
#pragma unroll
        for (int j = 0; j < 8; ++j)
          C[(size_t)(m0b + wm + tr * 16 + rb + j) * N +
            (n0b + wn + 16 * t + cl)] = acc[tr][t][j];
  }
}

// ---- RoPE (neox rotate-half) + q scale, f32 -> f16 ---------------------
__global__ void rope_kernel(const float* __restrict__ qf,
                            const float* __restrict__ kf,
                            half_t* __restrict__ qh, half_t* __restrict__ kh) {
  int p = blockIdx.x * blockDim.x + threadIdx.x;
  if (p >= MTOK * Hh * 32) return;
  int d = p & 31;
  int h = (p >> 5) & 15;
  int tok = p >> 9;
  int t = tok & (Tt - 1);
  float ang = (float)t * exp2f(-(float)d * 0.4152410118f);  // log2(1e4)/32
  float c = cosf(ang), s = sinf(ang);
  size_t base = (size_t)tok * Dd + h * DK + d;
  float q1 = qf[base], q2 = qf[base + 32];
  float k1 = kf[base], k2 = kf[base + 32];
  qh[base] = (half_t)((q1 * c - q2 * s) * 0.125f);
  qh[base + 32] = (half_t)((q1 * s + q2 * c) * 0.125f);
  kh[base] = (half_t)(k1 * c - k2 * s);
  kh[base + 32] = (half_t)(k1 * s + k2 * c);
}

// ---- chunked retention: one workgroup per (b,h) ------------------------
__global__ __launch_bounds__(256) void retention_kernel(
    const half_t* __restrict__ qh, const half_t* __restrict__ kh,
    const half_t* __restrict__ vh, float* __restrict__ o) {
  __shared__ half_t sQ[Cc * 72];    // q row-major [c][d]   (TDM target)
  __shared__ half_t sK[Cc * 72];    // k row-major [c][d]   (TDM target)
  __shared__ half_t sKt[DK * 72];   // (k*dkv)^T  [d][c]
  __shared__ half_t sVt[DV * 72];   // v^T        [e][c]
  __shared__ half_t sSt[DV * 72];   // state^T    [e][d]
  __shared__ half_t sA[Cc * 72];    // intra attention row-major [i][j]

  const int tid = threadIdx.x, lane = tid & 31, wid = tid >> 5;
  const int b = blockIdx.x >> 4, h = blockIdx.x & 15;
  const float gamma = 1.0f - exp2f(-5.0f - (float)h);
  const float lg = log2f(gamma);
  const float gC = exp2f(lg * (float)Cc);

  for (int i = tid; i < DV * 72; i += 256) sSt[i] = (half_t)0.0f;
  __syncthreads();

  const int m0 = (wid & 3) * 16;
  const int n0 = (wid >> 2) * 64;

  for (int n = 0; n < NC; ++n) {
    const size_t tok0 = (size_t)b * Tt + (size_t)n * Cc;
    {  // ---- stage q, k, k^T*dkv, v^T ----
      int r = tid >> 2;
      int c0 = (tid & 3) * 16;
      const half_t* ks = kh + (tok0 + r) * Dd + h * DK + c0;
      v8h kk0 = *(const v8h*)ks, kk1 = *(const v8h*)(ks + 8);
#ifdef USE_TDM
      if (tid < 32) {  // rows of 64 halves = 32 DWORDs -> pad code 4; +4 DW
        tdm_load_2d(qh + tok0 * Dd + h * DK, (unsigned)(size_t)sQ, 64, 64,
                    Dd, 4, 3);
        tdm_load_2d(kh + tok0 * Dd + h * DK, (unsigned)(size_t)sK, 64, 64,
                    Dd, 4, 3);
      }
#else
      const half_t* qs = qh + (tok0 + r) * Dd + h * DK + c0;
      v8h q0 = *(const v8h*)qs, q1 = *(const v8h*)(qs + 8);
      *(v8h*)&sQ[r * 72 + c0] = q0;
      *(v8h*)&sQ[r * 72 + c0 + 8] = q1;
      *(v8h*)&sK[r * 72 + c0] = kk0;
      *(v8h*)&sK[r * 72 + c0 + 8] = kk1;
#endif
      half_t dkv = (half_t)exp2f(lg * (float)(63 - r));
#pragma unroll
      for (int u = 0; u < 8; ++u) {
        sKt[(c0 + u) * 72 + r] = (half_t)(kk0[u] * dkv);
        sKt[(c0 + 8 + u) * 72 + r] = (half_t)(kk1[u] * dkv);
      }
      int c1 = (tid & 3) * 32;
      const half_t* vs = vh + (tok0 + r) * (2 * Dd) + h * DV + c1;
#pragma unroll
      for (int w = 0; w < 4; ++w) {
        v8h vv = *(const v8h*)(vs + 8 * w);
#pragma unroll
        for (int u = 0; u < 8; ++u) sVt[(c1 + 8 * w + u) * 72 + r] = vv[u];
      }
    }
    TDM_WAIT();
    __syncthreads();

    // ---- o_inter = (q*dq) @ S (old state via S^T) ----
    v8f oacc[4];
    {
      half_t dqrow = (half_t)exp2f(lg * (float)(m0 + (lane & 15) + 1));
      HV16 a0, a1;
      a0.v = frag_a(sQ, 72, m0, 0, lane);
      a1.v = frag_a(sQ, 72, m0, 32, lane);
#pragma unroll
      for (int i = 0; i < 16; ++i) {
        a0.v[i] = a0.v[i] * dqrow;
        a1.v[i] = a1.v[i] * dqrow;
      }
#pragma unroll
      for (int t = 0; t < 4; ++t) {
        oacc[t] = {};
        oacc[t] = wmma16(a0.v, frag_bT(sSt, 72, 0, n0 + 16 * t, lane), oacc[t]);
        oacc[t] = wmma16(a1.v, frag_bT(sSt, 72, 32, n0 + 16 * t, lane), oacc[t]);
      }
    }
    // ---- A = (q @ k^T) * Dmat; waves 0..3 own 16 rows each ----
    if (wid < 4) {
      int am0 = wid * 16;
      v16h a0 = frag_a(sQ, 72, am0, 0, lane);
      v16h a1 = frag_a(sQ, 72, am0, 32, lane);
#pragma unroll
      for (int t = 0; t < 4; ++t) {
        v8f acc = {};
        acc = wmma16(a0, frag_bT(sK, 72, 0, 16 * t, lane), acc);
        acc = wmma16(a1, frag_bT(sK, 72, 32, 16 * t, lane), acc);
        int colA = 16 * t + (lane & 15);
        int rbase = am0 + (lane >> 4) * 8;
#pragma unroll
        for (int j = 0; j < 8; ++j) {
          int rel = rbase + j - colA;
          float dm = (rel >= 0) ? exp2f(lg * (float)rel) : 0.0f;
          sA[(rbase + j) * 72 + colA] = (half_t)(acc[j] * dm);
        }
      }
    }
    __syncthreads();  // sA ready; all o_inter reads of S^T complete

    // ---- o_intra = A @ V (via V^T); store o ----
    {
      v16h a0 = frag_a(sA, 72, m0, 0, lane);
      v16h a1 = frag_a(sA, 72, m0, 32, lane);
#pragma unroll
      for (int t = 0; t < 4; ++t) {
        oacc[t] = wmma16(a0, frag_bT(sVt, 72, 0, n0 + 16 * t, lane), oacc[t]);
        oacc[t] = wmma16(a1, frag_bT(sVt, 72, 32, n0 + 16 * t, lane), oacc[t]);
      }
      int rb = (lane >> 4) * 8, cl = lane & 15;
#pragma unroll
      for (int t = 0; t < 4; ++t)
#pragma unroll
        for (int j = 0; j < 8; ++j)
          o[(tok0 + m0 + rb + j) * (2 * Dd) + h * DV + n0 + 16 * t + cl] =
              oacc[t][j];
    }
    // ---- S = gC*S + (k*dkv)^T @ V ----
    {
      v16h a0 = frag_a(sKt, 72, m0, 0, lane);
      v16h a1 = frag_a(sKt, 72, m0, 32, lane);
#pragma unroll
      for (int t = 0; t < 4; ++t) {
        v8f s = frag_cT_scaled(sSt, 72, m0, n0 + 16 * t, gC, lane);
        s = wmma16(a0, frag_bT(sVt, 72, 0, n0 + 16 * t, lane), s);
        s = wmma16(a1, frag_bT(sVt, 72, 32, n0 + 16 * t, lane), s);
        frag_cT_store(sSt, 72, m0, n0 + 16 * t, s, lane);
      }
    }
    __syncthreads();
  }
}

// ---- fused per-(b,t,h) RMSNorm + swish gate -> f16 ---------------------
__global__ __launch_bounds__(128) void gate_kernel(
    const float* __restrict__ o, const half_t* __restrict__ g,
    const float* __restrict__ gw, half_t* __restrict__ out) {
  __shared__ float red[128];
  int e = threadIdx.x;
  size_t base = (size_t)blockIdx.x * DV;
  float v = o[base + e];
  red[e] = v * v;
  __syncthreads();
#pragma unroll
  for (int s = 64; s > 0; s >>= 1) {
    if (e < s) red[e] += red[e + s];
    __syncthreads();
  }
  float r = rsqrtf(red[0] * (1.0f / 128.0f) + 1e-5f);
  float gv = (float)g[base + e];
  float sw = gv / (1.0f + expf(-gv));
  out[base + e] = (half_t)(v * r * gw[e] * sw);
}

extern "C" void kernel_launch(void* const* d_in, const int* in_sizes, int n_in,
                              void* d_out, int out_size, void* d_ws,
                              size_t ws_size, hipStream_t stream) {
  const float* x = (const float*)d_in[0];
  const float* Wq = (const float*)d_in[1];
  const float* Wk = (const float*)d_in[2];
  const float* Wv = (const float*)d_in[3];
  const float* Wg = (const float*)d_in[4];
  const float* Wo = (const float*)d_in[5];
  const float* gw = (const float*)d_in[6];
  float* out = (float*)d_out;

  char* wsb = (char*)d_ws;
  size_t off = 0;
  auto alloc = [&](size_t bytes) -> void* {
    void* p = wsb + off;
    off = (off + bytes + 255) & ~(size_t)255;
    return p;
  };
  half_t* xh  = (half_t*)alloc((size_t)MTOK * Dd * 2);
  half_t* Wqh = (half_t*)alloc((size_t)Dd * Dd * 2);
  half_t* Wkh = (half_t*)alloc((size_t)Dd * Dd * 2);
  half_t* Wvh = (half_t*)alloc((size_t)Dd * 2 * Dd * 2);
  half_t* Wgh = (half_t*)alloc((size_t)Dd * 2 * Dd * 2);
  half_t* Woh = (half_t*)alloc((size_t)2 * Dd * Dd * 2);
  float*  qf  = (float*)alloc((size_t)MTOK * Dd * 4);
  float*  kf  = (float*)alloc((size_t)MTOK * Dd * 4);
  half_t* vhh = (half_t*)alloc((size_t)MTOK * 2 * Dd * 2);
  half_t* ghh = (half_t*)alloc((size_t)MTOK * 2 * Dd * 2);
  half_t* qhh = (half_t*)alloc((size_t)MTOK * Dd * 2);
  half_t* khh = (half_t*)alloc((size_t)MTOK * Dd * 2);
  float*  oat = (float*)alloc((size_t)MTOK * 2 * Dd * 4);
  half_t* gat = (half_t*)alloc((size_t)MTOK * 2 * Dd * 2);

  auto cvt = [&](const float* src, half_t* dst, int n) {
    cvt_kernel<<<(n + 255) / 256, 256, 0, stream>>>(src, dst, n);
  };
  cvt(x, xh, MTOK * Dd);
  cvt(Wq, Wqh, Dd * Dd);
  cvt(Wk, Wkh, Dd * Dd);
  cvt(Wv, Wvh, Dd * 2 * Dd);
  cvt(Wg, Wgh, Dd * 2 * Dd);
  cvt(Wo, Woh, 2 * Dd * Dd);

  {
    dim3 blk(256);
    dim3 g1(Dd / 128, MTOK / 128);
    dim3 g2(2 * Dd / 128, MTOK / 128);
    gemm_f16_kernel<false><<<g1, blk, 0, stream>>>(xh, Wqh, qf, MTOK, Dd, Dd);
    gemm_f16_kernel<false><<<g1, blk, 0, stream>>>(xh, Wkh, kf, MTOK, Dd, Dd);
    gemm_f16_kernel<true><<<g2, blk, 0, stream>>>(xh, Wvh, vhh, MTOK, 2 * Dd, Dd);
    gemm_f16_kernel<true><<<g2, blk, 0, stream>>>(xh, Wgh, ghh, MTOK, 2 * Dd, Dd);
  }
  {
    int npairs = MTOK * Hh * 32;
    rope_kernel<<<(npairs + 255) / 256, 256, 0, stream>>>(qf, kf, qhh, khh);
  }
  retention_kernel<<<Bb * Hh, 256, 0, stream>>>(qhh, khh, vhh, oat);
  gate_kernel<<<MTOK * Hh, 128, 0, stream>>>(oat, ghh, gw, gat);
  {
    dim3 blk(256);
    dim3 g3(Dd / 128, MTOK / 128);
    gemm_f16_kernel<false><<<g3, blk, 0, stream>>>(gat, Woh, out, MTOK, Dd, 2 * Dd);
  }
}
